// CARAFEPack_81724637708599
// MI455X (gfx1250) — matrix operand order repaired and checked by reading, not compiled, not run on hardware
//
#include <hip/hip_runtime.h>
#include <hip/hip_bf16.h>
#include <math.h>

// ---------------- problem constants ----------------
constexpr int BATCH = 2;
constexpr int CH    = 128;
constexpr int OUTC  = 128;
constexpr int H     = 64, W = 64;
constexpr int K5    = 5;
constexpr int K2    = 25;
constexpr int COMP  = 64;
constexpr int NENC      = 100;   // K2*S*S
constexpr int NENC_PAD  = 128;   // padded to 8 n-tiles so every mode runs 4 tiles/wave
constexpr int KENC      = 576;   // 9 taps * 64 ch
constexpr int HS = 128, WS = 128;
constexpr int M1 = BATCH * H * W;     // 8192
constexpr int M3 = BATCH * HS * WS;   // 32768

typedef __bf16 bf16_t;
typedef __attribute__((ext_vector_type(16))) __bf16 v16bf;
typedef __attribute__((ext_vector_type(8)))  __bf16 v8bf;
typedef __attribute__((ext_vector_type(8)))  float  v8f;
typedef __attribute__((ext_vector_type(4)))  unsigned v4u;
typedef __attribute__((ext_vector_type(8)))  unsigned v8u;

union V16 { v16bf v; v8bf h[2]; };

__device__ __forceinline__ v8f wmma_bf16(v16bf a, v16bf b, v8f c) {
  return __builtin_amdgcn_wmma_f32_16x16x32_bf16(false, a, false, b,
                                                 (short)0, c, false, false);
}

// Issue one TDM copy: 64 rows x 32 k bf16 slab (row stride = kdim) -> LDS.
__device__ __forceinline__ void tdm_load_tile(const bf16_t* gptr, unsigned ldsOff,
                                              int kdim, int mtot) {
  unsigned long long ga = (unsigned long long)(uintptr_t)gptr;
  v4u g0;
  g0.x = 1u;                                               // count=1, no gather
  g0.y = ldsOff;                                           // lds_addr (bytes)
  g0.z = (unsigned)ga;                                     // global_addr[31:0]
  g0.w = (unsigned)((ga >> 32) & 0x1FFFFFFu) | (2u << 30); // addr[56:32], type=2
  v8u g1;
  g1[0] = (1u << 16);                                      // data_size = 2 bytes
  g1[1] = ((unsigned)kdim & 0xFFFFu) << 16;                // tensor_dim0 lo16
  g1[2] = (((unsigned)kdim >> 16) & 0xFFFFu) |
          (((unsigned)mtot & 0xFFFFu) << 16);              // dim0 hi16 | dim1 lo16
  g1[3] = (((unsigned)mtot >> 16) & 0xFFFFu) | (32u << 16);// dim1 hi16 | tile_dim0=32
  g1[4] = 64u;                                             // tile_dim1=64, tile_dim2=0
  g1[5] = (unsigned)kdim;                                  // tensor_dim0_stride lo32
  g1[6] = 0u;
  g1[7] = 0u;
  asm volatile("tensor_load_to_lds %0, %1" :: "s"(g0), "s"(g1) : "memory");
}

// ---------------- prep: fp32 -> bf16 repacks ----------------

// x (B,C,H,W) fp32 -> A1 (m = b*HW+hw, k = c) bf16
__global__ __launch_bounds__(256) void prep_x(const float* __restrict__ x,
                                              bf16_t* __restrict__ A1) {
  int i = blockIdx.x * 256 + threadIdx.x;
  if (i >= M1 * CH) return;
  int m = i >> 7;
  int c = i & 127;
  int b = m >> 12;
  int hw = m & 4095;
  A1[i] = (bf16_t)x[((b * CH + c) << 12) + hw];
}

// Weights -> operand-ready swizzled B tiles: for each (ktile, ntile),
// lane L: col n = ntile*16 + (L&15); elements e -> k = ktile*32 + (L>>4)*16 + e
__global__ __launch_bounds__(256) void prep_weights(const float* __restrict__ wc,
                                                    const float* __restrict__ we,
                                                    const float* __restrict__ wp,
                                                    bf16_t* __restrict__ W1S,
                                                    bf16_t* __restrict__ W2S,
                                                    bf16_t* __restrict__ W3S) {
  int i = blockIdx.x * 256 + threadIdx.x;
  // compress: K=128 (4 kt) x N=64 (4 nt)
  if (i < 4 * 4 * 512) {
    int e = i & 15, lane = (i >> 4) & 31, nt = (i >> 9) & 3, kt = i >> 11;
    int n = nt * 16 + (lane & 15);
    int k = kt * 32 + (lane >> 4) * 16 + e;
    W1S[i] = (bf16_t)wc[n * CH + k];
  }
  // encoder: K=576 (18 kt) x Npad=128 (8 nt); k = tap*64 + c ordering
  if (i < 18 * 8 * 512) {
    int e = i & 15, lane = (i >> 4) & 31;
    int t2 = i >> 9;
    int nt = t2 & 7, kt = t2 >> 3;
    int n = nt * 16 + (lane & 15);
    int k = kt * 32 + (lane >> 4) * 16 + e;
    int tap = k >> 6, c = k & 63;
    float v = (n < NENC) ? we[(n * COMP + c) * 9 + tap] : 0.f;
    W2S[i] = (bf16_t)v;
  }
  // projection: K=128 (4 kt) x N=128 (8 nt)
  if (i < 4 * 8 * 512) {
    int e = i & 15, lane = (i >> 4) & 31, nt = (i >> 9) & 7, kt = i >> 12;
    int n = nt * 16 + (lane & 15);
    int k = kt * 32 + (lane >> 4) * 16 + e;
    W3S[i] = (bf16_t)wp[n * CH + k];
  }
}

// ---------------- WMMA GEMM: 64M x 64N per block, 4 WMMAs per wave per K-step,
// double-buffered LDS A tile fed by TDM (plain) or async copies (im2col) ----------------
constexpr int MODE_COMPRESS = 0;
constexpr int MODE_ENCODER  = 1;
constexpr int MODE_PROJ     = 2;

template <int MODE>
__global__ __launch_bounds__(128)
void gemm_wmma(const bf16_t* __restrict__ A,    // (M,K) bf16, or compressed NHWC for ENCODER
               const bf16_t* __restrict__ Bsw,  // swizzled B tiles (kt, nt, lane, 16)
               const float*  __restrict__ bias,
               float* __restrict__ outF,
               bf16_t* __restrict__ outB) {
  constexpr int KDIM   = (MODE == MODE_ENCODER) ? KENC : CH;
  constexpr int NTILES = (MODE == MODE_COMPRESS) ? (COMP / 16) : 8;
  constexpr int MTOT   = (MODE == MODE_PROJ) ? M3 : M1;
  constexpr int KSTEPS = KDIM / 32;

  __shared__ bf16_t As[2][64 * 32];   // double-buffered A tile (4 KB each)

  const int tid   = threadIdx.x;
  const int lane  = tid & 31;        // wave32
  const int wv    = tid >> 5;
  const int half  = lane >> 4;
  const int l15   = lane & 15;
  const int mBase = blockIdx.x * 64;
  const int ntBase = blockIdx.y * 4; // 4 n-tiles per wave

  // async im2col staging for one K-slab into buffer `buf`
  auto stage_async = [&](int k0, int buf) {
    const int tap = k0 >> 6;
    const int c0  = k0 & 63;
    const int dy = tap / 3 - 1, dx = tap % 3 - 1;
    for (int t = tid; t < 64 * 4; t += 128) {
      int row = t >> 2;
      int kc  = (t & 3) * 8;
      int m = mBase + row;
      int b = m >> 12;
      int hw = m & 4095;
      int hh = (hw >> 6) + dy;
      int ww = (hw & 63) + dx;
      unsigned ldsA = (unsigned)(uintptr_t)&As[buf][row * 32 + kc];
      if ((unsigned)hh < 64u && (unsigned)ww < 64u) {
        unsigned long long ga = (unsigned long long)(uintptr_t)
            &A[(size_t)(((b << 12) + (hh << 6) + ww) << 6) + c0 + kc];
        asm volatile("global_load_async_to_lds_b128 %0, %1, off"
                     :: "v"(ldsA), "v"(ga) : "memory");
      } else {
        uint4 z = {0u, 0u, 0u, 0u};
        *(uint4*)&As[buf][row * 32 + kc] = z;    // zero-fill conv halo
      }
    }
  };

  // accumulators: 4 n-tiles, bias-initialized
  v8f acc[4];
#pragma unroll
  for (int j = 0; j < 4; ++j) {
    int n = (ntBase + j) * 16 + l15;
    float bn = 0.f;
    if constexpr (MODE == MODE_ENCODER) { if (n < NENC) bn = bias[n]; }
    else bn = bias[n];
#pragma unroll
    for (int r = 0; r < 8; ++r) acc[j][r] = bn;
  }

  // ---- preload K-slab 0
  if constexpr (MODE == MODE_ENCODER) {
    stage_async(0, 0);
  } else {
    if (tid == 0)
      tdm_load_tile(&A[(size_t)mBase * KDIM], (unsigned)(uintptr_t)&As[0][0],
                    KDIM, MTOT);
  }

  for (int it = 0; it < KSTEPS; ++it) {
    const int buf = it & 1;
    // ---- wait for slab `it`, publish to the workgroup
    if constexpr (MODE == MODE_ENCODER) {
      asm volatile("s_wait_asynccnt 0x0" ::: "memory");
    } else {
      if (tid == 0) __builtin_amdgcn_s_wait_tensorcnt(0);
    }
    __syncthreads();
    // ---- kick off slab `it+1` into the other buffer (overlaps with WMMAs below)
    if (it + 1 < KSTEPS) {
      if constexpr (MODE == MODE_ENCODER) {
        stage_async((it + 1) * 32, buf ^ 1);
      } else {
        if (tid == 0)
          tdm_load_tile(&A[(size_t)mBase * KDIM + (it + 1) * 32],
                        (unsigned)(uintptr_t)&As[buf ^ 1][0], KDIM, MTOT);
        __builtin_prefetch(
            &Bsw[((size_t)((it + 1) * NTILES + ntBase) * 32 + lane) * 16], 0, 1);
      }
    }
    // ---- 4 WMMAs: one A operand reused across 4 B tiles
    V16 a;
    a.h[0] = *(const v8bf*)&As[buf][(wv * 16 + l15) * 32 + half * 16];
    a.h[1] = *(const v8bf*)&As[buf][(wv * 16 + l15) * 32 + half * 16 + 8];
#pragma unroll
    for (int j = 0; j < 4; ++j) {
      v16bf b = *(const v16bf*)
          &Bsw[((size_t)(it * NTILES + ntBase + j) * 32 + lane) * 16];
      acc[j] = wmma_bf16(a.v, b, acc[j]);
    }
    __syncthreads();   // all waves done reading buf before it is overwritten
  }

  // ---- store D: lane col = l15; VGPR r -> row 8*half + r
#pragma unroll
  for (int j = 0; j < 4; ++j) {
    int n = (ntBase + j) * 16 + l15;
#pragma unroll
    for (int r = 0; r < 8; ++r) {
      int m = mBase + wv * 16 + half * 8 + r;
      float v = acc[j][r];
      if constexpr (MODE == MODE_COMPRESS) {
        outB[(size_t)m * COMP + n] = (bf16_t)v;             // compressed NHWC bf16
      } else if constexpr (MODE == MODE_ENCODER) {
        if (n < NENC) outF[(size_t)m * NENC + n] = v;       // enc NHWC fp32
      } else {
        int b   = m >> 14;
        int pix = m & 16383;
        outF[((size_t)(b * OUTC + n) << 14) + pix] = v;     // (B, OUT, HS, WS) fp32
      }
    }
  }
}

// ---------------- pixel-shuffle + softmax over 25 taps ----------------
__global__ __launch_bounds__(256) void shuffle_softmax(const float* __restrict__ enc,
                                                       float* __restrict__ kn) {
  int i = blockIdx.x * 256 + threadIdx.x;
  if (i >= BATCH * HS * WS) return;
  int b  = i >> 14;
  int hs = (i >> 7) & 127;
  int ws = i & 127;
  int h = hs >> 1, w = ws >> 1;
  int sub = ((hs & 1) << 1) | (ws & 1);
  const float* p = &enc[(size_t)((b << 12) + (h << 6) + w) * NENC + sub];
  float vals[K2];
  float mx = -1e30f;
#pragma unroll
  for (int k = 0; k < K2; ++k) { vals[k] = p[k * 4]; mx = fmaxf(mx, vals[k]); }
  float s = 0.f;
#pragma unroll
  for (int k = 0; k < K2; ++k) { vals[k] = __expf(vals[k] - mx); s += vals[k]; }
  float inv = 1.f / s;
  float* o = &kn[(size_t)i * K2];
#pragma unroll
  for (int k = 0; k < K2; ++k) o[k] = vals[k] * inv;
}

// ---------------- content-aware reassembly (5x5 stencil in half-res space) ----------------
__global__ __launch_bounds__(256) void reassemble(const float* __restrict__ x,
                                                  const float* __restrict__ kn,
                                                  bf16_t* __restrict__ R) {
  int i = blockIdx.x * 256 + threadIdx.x;
  int ws = i & 127;
  int hs = (i >> 7) & 127;
  int c  = (i >> 14) & 127;
  int b  = i >> 21;
  int h0 = (hs >> 1) - 2;
  int w0 = (ws >> 1) - 2;
  const float* kp = &kn[(size_t)((b << 14) + (hs << 7) + ws) * K2];
  const float* xb = &x[(size_t)((b * CH + c) << 12)];
  float acc = 0.f;
#pragma unroll
  for (int iy = 0; iy < K5; ++iy) {
    int y = h0 + iy;
    bool yok = (unsigned)y < 64u;
#pragma unroll
    for (int jx = 0; jx < K5; ++jx) {
      int xx = w0 + jx;
      float xv = (yok && (unsigned)xx < 64u) ? xb[(y << 6) + xx] : 0.f;
      acc = fmaf(kp[iy * K5 + jx], xv, acc);
    }
  }
  R[((size_t)((b << 14) + (hs << 7) + ws) << 7) + c] = (bf16_t)acc;
}

// ---------------- host: pipeline launch ----------------
extern "C" void kernel_launch(void* const* d_in, const int* in_sizes, int n_in,
                              void* d_out, int out_size, void* d_ws, size_t ws_size,
                              hipStream_t stream) {
  const float* x      = (const float*)d_in[0];
  const float* w_comp = (const float*)d_in[1];
  const float* b_comp = (const float*)d_in[2];
  const float* w_enc  = (const float*)d_in[3];
  const float* b_enc  = (const float*)d_in[4];
  const float* w_proj = (const float*)d_in[5];
  const float* b_proj = (const float*)d_in[6];
  float* out = (float*)d_out;

  char* ws = (char*)d_ws;
  size_t off = 0;
  auto carve = [&](size_t bytes) {
    char* p = ws + off;
    off = (off + bytes + 255) & ~(size_t)255;
    return p;
  };
  bf16_t* A1   = (bf16_t*)carve((size_t)M1 * CH * 2);       // 2 MB
  bf16_t* W1S  = (bf16_t*)carve((size_t)4 * 4 * 512 * 2);   // 16 KB  swizzled B (compress)
  bf16_t* Comp = (bf16_t*)carve((size_t)M1 * COMP * 2);     // 1 MB
  bf16_t* W2S  = (bf16_t*)carve((size_t)18 * 8 * 512 * 2);  // 144 KB swizzled B (encoder)
  float*  Enc  = (float*) carve((size_t)M1 * NENC * 4);     // 3.1 MB
  float*  Kn   = (float*) carve((size_t)M3 * K2 * 4);       // 3.1 MB
  bf16_t* Rs   = (bf16_t*)carve((size_t)M3 * CH * 2);       // 8 MB
  bf16_t* W3S  = (bf16_t*)carve((size_t)4 * 8 * 512 * 2);   // 32 KB  swizzled B (proj)
  if (off > ws_size) return;

  prep_x<<<(M1 * CH) / 256, 256, 0, stream>>>(x, A1);
  prep_weights<<<(18 * 8 * 512 + 255) / 256, 256, 0, stream>>>(
      w_comp, w_enc, w_proj, W1S, W2S, W3S);

  // compress: 64N in one block column
  gemm_wmma<MODE_COMPRESS><<<dim3(M1 / 64, 1), 128, 0, stream>>>(
      A1, W1S, b_comp, nullptr, Comp);

  // encoder: Npad=128 -> 2 block columns
  gemm_wmma<MODE_ENCODER><<<dim3(M1 / 64, 2), 128, 0, stream>>>(
      Comp, W2S, b_enc, Enc, nullptr);

  shuffle_softmax<<<(BATCH * HS * WS) / 256, 256, 0, stream>>>(Enc, Kn);

  reassemble<<<(BATCH * CH * HS * WS) / 256, 256, 0, stream>>>(x, Kn, Rs);

  // projection: N=128 -> 2 block columns
  gemm_wmma<MODE_PROJ><<<dim3(M3 / 64, 2), 128, 0, stream>>>(
      Rs, W3S, b_proj, out, nullptr);
}
